// MultiHeadAttention_5076651344343
// MI455X (gfx1250) — matrix operand order repaired and checked by reading, not compiled, not run on hardware
//
#include <hip/hip_runtime.h>
#include <hip/hip_bf16.h>

// ---------------------------------------------------------------------------
// MHA forward for MI455X (gfx1250, wave32, WMMA 16x16x32 f16->f32)
// Shapes: B=2, S=2048, D=1024, H=16, DH=64
// Register-blocked GEMMs (4Mx2N, double-buffered k-loop) +
// 2-query-tile flash attention. __launch_bounds__(128,1) lifts the VGPR cap
// so the ~200-VGPR double-buffered tiles allocate without scratch spills.
// ---------------------------------------------------------------------------

typedef __attribute__((ext_vector_type(16))) _Float16 v16h;
typedef __attribute__((ext_vector_type(8)))  _Float16 v8h;
typedef __attribute__((ext_vector_type(8)))  float    v8f;
typedef __attribute__((ext_vector_type(4)))  float    v4f;

constexpr int NB   = 2;
constexpr int NS   = 2048;
constexpr int ND   = 1024;
constexpr int NH   = 16;
constexpr int NDH  = 64;
constexpr int NTOK = NB * NS;      // 4096
constexpr int WPB  = 4;            // waves per 128-thread block (wave32)

#define WMMA_F16(a, b, c) \
  __builtin_amdgcn_wmma_f32_16x16x32_f16(false, (a), false, (b), (short)0, (c), false, false)

// ---- fragment loaders (k order k = 16*half + s, same for A and B) ----------
// A 16x32 f16: lane L(0-15)=row M=L holds K=0..15; lane L+16 same
// row, K=16..31. Two contiguous 8-f16 (16B) chunks per lane.
__device__ __forceinline__ v16h load_a16(const _Float16* rowk, int lhalf) {
  const v8h a0 = *(const v8h*)(rowk + 16 * lhalf);
  const v8h a1 = *(const v8h*)(rowk + 16 * lhalf + 8);
  v16h r;
#pragma unroll
  for (int i = 0; i < 8; ++i) { r[i] = a0[i]; r[i + 8] = a1[i]; }
  return r;
}
// B 32x16 f16: lane n(0-15)=col holds K=0..15, lane n+16 holds K=16..31.
// One contiguous 16-f16 (32B) chunk per lane.
__device__ __forceinline__ v16h load_b16(const _Float16* rowk, int lhalf) {
  return *(const v16h*)(rowk + 16 * lhalf);
}

// ---------------------------------------------------------------------------
// fp32 -> fp16 conversion (grid-stride)
// ---------------------------------------------------------------------------
__global__ void cvt_f32_f16(const float* __restrict__ src,
                            _Float16* __restrict__ dst, int n) {
  int i = blockIdx.x * blockDim.x + threadIdx.x;
  int stride = gridDim.x * blockDim.x;
  for (; i < n; i += stride) dst[i] = (_Float16)src[i];
}

// ---------------------------------------------------------------------------
// Double-buffered 4Mx2N WMMA GEMM core:
//   c[mi][ni] += sum_k A[mi-rows, k] * B[ni-rows, k]^T  (both k-contiguous)
// Loads for slice k+32 issue while WMMAs for slice k execute.
// ---------------------------------------------------------------------------
template <int BM, int BN>
__device__ __forceinline__ void gemm_core(const _Float16* const (&arow)[BM],
                                          const _Float16* const (&brow)[BN],
                                          int kdim, int lhalf,
                                          v8f (&c)[BM][BN]) {
  v16h a0[BM], b0[BN], a1[BM], b1[BN];
#pragma unroll
  for (int mi = 0; mi < BM; ++mi) a0[mi] = load_a16(arow[mi], lhalf);
#pragma unroll
  for (int ni = 0; ni < BN; ++ni) b0[ni] = load_b16(brow[ni], lhalf);

  for (int k = 0; k < kdim; k += 64) {
    __builtin_prefetch(arow[0] + k + 128, 0, 1);
    __builtin_prefetch(brow[0] + k + 128, 0, 1);
    // stage k+32 into buffer 1
#pragma unroll
    for (int mi = 0; mi < BM; ++mi) a1[mi] = load_a16(arow[mi] + k + 32, lhalf);
#pragma unroll
    for (int ni = 0; ni < BN; ++ni) b1[ni] = load_b16(brow[ni] + k + 32, lhalf);
    // compute slice k (buffer 0)
#pragma unroll
    for (int mi = 0; mi < BM; ++mi)
#pragma unroll
      for (int ni = 0; ni < BN; ++ni)
        c[mi][ni] = WMMA_F16(a0[mi], b0[ni], c[mi][ni]);
    // stage k+64 into buffer 0 (skipped on last iteration)
    if (k + 64 < kdim) {
#pragma unroll
      for (int mi = 0; mi < BM; ++mi) a0[mi] = load_a16(arow[mi] + k + 64, lhalf);
#pragma unroll
      for (int ni = 0; ni < BN; ++ni) b0[ni] = load_b16(brow[ni] + k + 64, lhalf);
    }
    // compute slice k+32 (buffer 1)
#pragma unroll
    for (int mi = 0; mi < BM; ++mi)
#pragma unroll
      for (int ni = 0; ni < BN; ++ni)
        c[mi][ni] = WMMA_F16(a1[mi], b1[ni], c[mi][ni]);
  }
}

// ---------------------------------------------------------------------------
// QKV projection, 4Mx2N per wave:
//   C^T[dout, tok] = sum_k Wi[dout,k] * X[tok,k] + bi[dout]
// Writes Q,K as (B,H,S,DH) f16 and V transposed as (B,H,DH,S) f16.
// ---------------------------------------------------------------------------
__global__ __launch_bounds__(128, 1)
void qkv_gemm(const _Float16* __restrict__ X,
              const _Float16* __restrict__ Wi,
              const float* __restrict__ bi,
              _Float16* __restrict__ Q,
              _Float16* __restrict__ K,
              _Float16* __restrict__ VT) {
  constexpr int BM = 4, BN = 2;
  constexpr int MTm = (3 * ND / 16) / BM;  // 48
  constexpr int NTm = (NTOK / 16) / BN;    // 128
  const int lane = threadIdx.x & 31;
  const int ln = lane & 15;
  const int lhalf = lane >> 4;
  const int wave = blockIdx.x * WPB + (threadIdx.x >> 5);
  if (wave >= MTm * NTm) return;
  const int mtm = wave % MTm;
  const int ntm = wave / MTm;

  const _Float16* arow[BM];
  const _Float16* brow[BN];
#pragma unroll
  for (int mi = 0; mi < BM; ++mi)
    arow[mi] = Wi + (size_t)(mtm * 64 + mi * 16 + ln) * ND;
#pragma unroll
  for (int ni = 0; ni < BN; ++ni)
    brow[ni] = X + (size_t)(ntm * 32 + ni * 16 + ln) * ND;

  v8f c[BM][BN];
#pragma unroll
  for (int mi = 0; mi < BM; ++mi)
#pragma unroll
    for (int ni = 0; ni < BN; ++ni)
      c[mi][ni] = (v8f){0.f,0.f,0.f,0.f,0.f,0.f,0.f,0.f};

  gemm_core<BM, BN>(arow, brow, ND, lhalf, c);

#pragma unroll
  for (int mi = 0; mi < BM; ++mi) {
    const int dout0 = mtm * 64 + mi * 16 + 8 * lhalf;  // 8 consecutive channels
    const int sel  = dout0 >> 10;                      // 0=Q 1=K 2=V
    const int dmod = dout0 & (ND - 1);
    const int h    = dmod >> 6;
    const int dh0  = dmod & (NDH - 1);
#pragma unroll
    for (int ni = 0; ni < BN; ++ni) {
      const int tok = ntm * 32 + ni * 16 + ln;
      const int bb = tok >> 11;
      const int s  = tok & (NS - 1);
      if (sel == 2) {
        // V stored transposed: VT[(bh*DH + dh)*S + s]
#pragma unroll
        for (int r = 0; r < 8; ++r) {
          float v = c[mi][ni][r] + bi[dout0 + r];
          VT[((size_t)(bb * NH + h) * NDH + dh0 + r) * NS + s] = (_Float16)v;
        }
      } else {
        v8h pk;
#pragma unroll
        for (int r = 0; r < 8; ++r) pk[r] = (_Float16)(c[mi][ni][r] + bi[dout0 + r]);
        _Float16* dst = (sel == 0 ? Q : K) +
                        ((size_t)(bb * NH + h) * NS + s) * NDH + dh0;
        *(v8h*)dst = pk;   // 16B packed store
      }
    }
  }
}

// ---------------------------------------------------------------------------
// Online-softmax update for one 16-query tile over a 32-key block.
// c0/c1: scores^T C-tiles (keys kb..kb+15 / kb+16..kb+31 in VGPRs, query=lane).
// Updates running max/sum, rescales out accumulators, returns P^T B-fragment.
// ---------------------------------------------------------------------------
__device__ __forceinline__ v16h softmax_update(
    const v8f& c0, const v8f& c1, int kb, int q, int lhalf, int causal,
    float& mrun, float& lrun, v8f* o /* 4 d-tiles */) {
  const float scale = 0.125f;                   // 1/sqrt(64)
  float s0[8], s1[8];
  float mloc = -1e30f;
#pragma unroll
  for (int r = 0; r < 8; ++r) {
    const int kk0 = kb + r + 8 * lhalf;
    const int kk1 = kk0 + 16;
    float x0 = c0[r] * scale;
    float x1 = c1[r] * scale;
    if (causal && kk0 > q) x0 = -1e30f;
    if (causal && kk1 > q) x1 = -1e30f;
    s0[r] = x0; s1[r] = x1;
    mloc = fmaxf(mloc, fmaxf(x0, x1));
  }
  // lane q holds half the keys, lane q^16 the other half
  mloc = fmaxf(mloc, __shfl_xor(mloc, 16, 32));
  const float mnew = fmaxf(mrun, mloc);
  const float corr = __expf(mrun - mnew);
  mrun = mnew;
  lrun *= corr;
#pragma unroll
  for (int t = 0; t < 4; ++t)
#pragma unroll
    for (int r = 0; r < 8; ++r) o[t][r] *= corr;

  float p0[8], p1[8], ls = 0.f;
#pragma unroll
  for (int r = 0; r < 8; ++r) {
    p0[r] = __expf(s0[r] - mnew);
    p1[r] = __expf(s1[r] - mnew);
    ls += p0[r] + p1[r];
  }
  ls += __shfl_xor(ls, 16, 32);
  lrun += ls;

  float t0[8], t1[8];
#pragma unroll
  for (int r = 0; r < 8; ++r) {
    t0[r] = __shfl_xor(p0[r], 16, 32);
    t1[r] = __shfl_xor(p1[r], 16, 32);
  }
  v16h pb;
#pragma unroll
  for (int j = 0; j < 8; ++j) {
    pb[j]     = (_Float16)(lhalf ? t1[j] : p0[j]);  // K = 16*half + j
    pb[8 + j] = (_Float16)(lhalf ? p1[j] : t0[j]);  // K = 16*half + 8 + j
  }
  return pb;
}

// ---------------------------------------------------------------------------
// Flash attention, one wave per 32-query macro tile per (b,h).
// scores^T = K_tile(16k x 64d) @ Q^T(64d x 16q): keys in VGPRs, queries in
// lanes -> per-lane online softmax + one xor-16 merge. K and V A-fragments
// are reused across both query tiles; V loads hoisted ahead of softmax so
// they are in flight while shuffles/exps execute.
// ---------------------------------------------------------------------------
__global__ __launch_bounds__(128, 1)
void flash_attn(const _Float16* __restrict__ Q,
                const _Float16* __restrict__ K,
                const _Float16* __restrict__ VT,
                const int* __restrict__ causal_p,
                _Float16* __restrict__ O) {
  const int lane = threadIdx.x & 31;
  const int ln = lane & 15;
  const int lhalf = lane >> 4;
  const int wave = blockIdx.x * WPB + (threadIdx.x >> 5);
  constexpr int QT = NS / 32;                  // 64 query macro tiles
  if (wave >= NB * NH * QT) return;
  const int bh = wave / QT;
  const int qb = (wave % QT) * 32;
  const int causal = causal_p[0];

  const _Float16* Qb = Q  + (size_t)bh * NS * NDH;
  const _Float16* Kb = K  + (size_t)bh * NS * NDH;
  const _Float16* Vb = VT + (size_t)bh * NDH * NS;

  // Preload Q^T B-fragments: 2 query tiles x 2 dh-chunks (lane = query)
  v16h bq[2][2];
#pragma unroll
  for (int q2 = 0; q2 < 2; ++q2) {
    const _Float16* qrow = Qb + (size_t)(qb + q2 * 16 + ln) * NDH;
    bq[q2][0] = load_b16(qrow, lhalf);
    bq[q2][1] = load_b16(qrow + 32, lhalf);
  }

  v8f o[2][4];
#pragma unroll
  for (int q2 = 0; q2 < 2; ++q2)
#pragma unroll
    for (int t = 0; t < 4; ++t)
      o[q2][t] = (v8f){0.f,0.f,0.f,0.f,0.f,0.f,0.f,0.f};
  float mrun[2] = {-1e30f, -1e30f};
  float lrun[2] = {0.f, 0.f};
  const int kend = causal ? (qb + 32) : NS;

  for (int kb = 0; kb < kend; kb += 32) {
    // ---- K A-fragments (2 key tiles x 2 dh-chunks) + V A-fragments ----
    v16h ak[2][2];
#pragma unroll
    for (int kt = 0; kt < 2; ++kt) {
      const _Float16* krow = Kb + (size_t)(kb + kt * 16 + ln) * NDH;
      ak[kt][0] = load_a16(krow, lhalf);
      ak[kt][1] = load_a16(krow + 32, lhalf);
    }
    v16h av[4];
#pragma unroll
    for (int t = 0; t < 4; ++t)
      av[t] = load_a16(Vb + (size_t)(t * 16 + ln) * NS + kb, lhalf);

    // ---- scores^T: 4 C tiles (2 query tiles x 2 key tiles), K-dim=64 ----
    v8f c[2][2];
#pragma unroll
    for (int q2 = 0; q2 < 2; ++q2)
#pragma unroll
      for (int kt = 0; kt < 2; ++kt) {
        v8f acc = (v8f){0.f,0.f,0.f,0.f,0.f,0.f,0.f,0.f};
        acc = WMMA_F16(ak[kt][0], bq[q2][0], acc);
        acc = WMMA_F16(ak[kt][1], bq[q2][1], acc);
        c[q2][kt] = acc;
      }

    // ---- per-query-tile online softmax + P^T fragment ----
    v16h pb[2];
#pragma unroll
    for (int q2 = 0; q2 < 2; ++q2)
      pb[q2] = softmax_update(c[q2][0], c[q2][1], kb, qb + q2 * 16 + ln,
                              lhalf, causal, mrun[q2], lrun[q2], o[q2]);

    // ---- out^T += V^T(16d x 32k) @ P^T : V fragment shared by query tiles
#pragma unroll
    for (int t = 0; t < 4; ++t)
#pragma unroll
      for (int q2 = 0; q2 < 2; ++q2)
        o[q2][t] = WMMA_F16(av[t], pb[q2], o[q2][t]);
  }

  // ---- normalize and store: lane = query, regs = 8 consecutive d-channels
  const int b = bh >> 4;                 // / NH
  const int h = bh & 15;
#pragma unroll
  for (int q2 = 0; q2 < 2; ++q2) {
    const float inv = 1.0f / lrun[q2];
    const size_t tokrow = (size_t)(b * NS + qb + q2 * 16 + ln) * ND;
#pragma unroll
    for (int t = 0; t < 4; ++t) {
      v8h pk;
#pragma unroll
      for (int r = 0; r < 8; ++r) pk[r] = (_Float16)(o[q2][t][r] * inv);
      const int col0 = h * NDH + t * 16 + 8 * lhalf;
      *(v8h*)(O + tokrow + col0) = pk;   // 16B packed store
    }
  }
}

// ---------------------------------------------------------------------------
// Output projection, 4Mx2N, double-buffered:
//   out[tok, dout] = sum_k Attn[tok,k]*Wo[dout,k] + bo[dout]   (f32 output)
// ---------------------------------------------------------------------------
__global__ __launch_bounds__(128, 1)
void out_gemm(const _Float16* __restrict__ Attn,
              const _Float16* __restrict__ Wo,
              const float* __restrict__ bo,
              float* __restrict__ out) {
  constexpr int BM = 4, BN = 2;
  constexpr int MTm = (ND / 16) / BM;      // 16
  constexpr int NTm = (NTOK / 16) / BN;    // 128
  const int lane = threadIdx.x & 31;
  const int ln = lane & 15;
  const int lhalf = lane >> 4;
  const int wave = blockIdx.x * WPB + (threadIdx.x >> 5);
  if (wave >= MTm * NTm) return;
  const int mtm = wave % MTm;
  const int ntm = wave / MTm;

  const _Float16* arow[BM];
  const _Float16* brow[BN];
#pragma unroll
  for (int mi = 0; mi < BM; ++mi)
    arow[mi] = Wo + (size_t)(mtm * 64 + mi * 16 + ln) * ND;
#pragma unroll
  for (int ni = 0; ni < BN; ++ni)
    brow[ni] = Attn + (size_t)(ntm * 32 + ni * 16 + ln) * ND;

  v8f c[BM][BN];
#pragma unroll
  for (int mi = 0; mi < BM; ++mi)
#pragma unroll
    for (int ni = 0; ni < BN; ++ni)
      c[mi][ni] = (v8f){0.f,0.f,0.f,0.f,0.f,0.f,0.f,0.f};

  gemm_core<BM, BN>(arow, brow, ND, lhalf, c);

#pragma unroll
  for (int mi = 0; mi < BM; ++mi) {
    const int dout0 = mtm * 64 + mi * 16 + 8 * lhalf;
#pragma unroll
    for (int ni = 0; ni < BN; ++ni) {
      const int tok = ntm * 32 + ni * 16 + ln;
      v4f lo, hi;
#pragma unroll
      for (int r = 0; r < 4; ++r) {
        lo[r] = c[mi][ni][r]     + bo[dout0 + r];
        hi[r] = c[mi][ni][4 + r] + bo[dout0 + 4 + r];
      }
      float* dst = out + (size_t)tok * ND + dout0;   // 8 consecutive f32
      *(v4f*)dst = lo;
      *(v4f*)(dst + 4) = hi;
    }
  }
}

// ---------------------------------------------------------------------------
// Launch
// ---------------------------------------------------------------------------
extern "C" void kernel_launch(void* const* d_in, const int* in_sizes, int n_in,
                              void* d_out, int out_size, void* d_ws, size_t ws_size,
                              hipStream_t stream) {
  const float* x      = (const float*)d_in[0];   // (2,2048,1024)
  const float* Wi     = (const float*)d_in[1];   // (3072,1024)
  const float* bi     = (const float*)d_in[2];   // (3072)
  const float* Wo     = (const float*)d_in[3];   // (1024,1024)
  const float* bo     = (const float*)d_in[4];   // (1024)
  const int*   causal = (const int*)d_in[5];     // scalar
  float* out = (float*)d_out;                    // (2,2048,1024) f32

  char* ws = (char*)d_ws;
  constexpr size_t SZ_XH  = (size_t)NTOK * ND * 2;        //  8 MB
  constexpr size_t SZ_WIH = (size_t)3 * ND * ND * 2;      //  6 MB
  constexpr size_t SZ_WOH = (size_t)ND * ND * 2;          //  2 MB
  constexpr size_t SZ_HD  = (size_t)NTOK * ND * 2;        //  8 MB each
  _Float16* Xh   = (_Float16*)(ws);
  _Float16* Wih  = (_Float16*)(ws + SZ_XH);
  _Float16* Woh  = (_Float16*)(ws + SZ_XH + SZ_WIH);
  _Float16* Qh   = (_Float16*)(ws + SZ_XH + SZ_WIH + SZ_WOH);
  _Float16* Kh   = (_Float16*)(ws + SZ_XH + SZ_WIH + SZ_WOH + SZ_HD);
  _Float16* VTh  = (_Float16*)(ws + SZ_XH + SZ_WIH + SZ_WOH + 2 * SZ_HD);
  _Float16* Atth = (_Float16*)(ws + SZ_XH + SZ_WIH + SZ_WOH + 3 * SZ_HD);

  // 1) f32 -> f16 conversions
  cvt_f32_f16<<<2048, 256, 0, stream>>>(x,  Xh,  NTOK * ND);
  cvt_f32_f16<<<2048, 256, 0, stream>>>(Wi, Wih, 3 * ND * ND);
  cvt_f32_f16<<<1024, 256, 0, stream>>>(Wo, Woh, ND * ND);

  // 2) QKV projection: 48 x 128 macro tiles, 4 waves/block of 128
  qkv_gemm<<<(48 * 128) / WPB, 128, 0, stream>>>(Xh, Wih, bi, Qh, Kh, VTh);

  // 3) flash attention: B*H*(S/32) = 2048 waves
  flash_attn<<<(NB * NH * (NS / 32)) / WPB, 128, 0, stream>>>(Qh, Kh, VTh, causal, Atth);

  // 4) output projection: 16 x 128 macro tiles
  out_gemm<<<(16 * 128) / WPB, 128, 0, stream>>>(Atth, Woh, bo, out);
}
